// _CoreAttention_5686536700528
// MI455X (gfx1250) — compile-verified, matrix operand order checked by reading, not run on hardware
//
#include <hip/hip_runtime.h>
#include <hip/hip_bf16.h>

// ---------------- types ----------------
typedef _Float16 h16 __attribute__((ext_vector_type(16)));
typedef _Float16 h8  __attribute__((ext_vector_type(8)));
typedef _Float16 h4  __attribute__((ext_vector_type(4)));
typedef float    f8  __attribute__((ext_vector_type(8)));

// Problem constants
#define NW   64     // window tokens (padded)
#define NA   60     // actual tokens
#define CDIM 512
#define NH   16
#define HD   32
#define C3   1536

// LDS strides (in halves) — padded so 16-lane row accesses spread banks
#define XS   520    // xh / outh stride  (64 rows)
#define QKS  1032   // q|k buffer stride (64 rows, 1024 cols)
#define VTS  72     // vT stride (512 rows, 64 cols)
#define PS   72     // per-wave P buffer stride (16 rows)

// LDS offsets in halves
#define OFF_XH  0
#define OFF_QK  (64 * XS)                  // 33280
#define OFF_VT  (OFF_QK + 64 * QKS)        // 99328
#define OFF_PB  (OFF_VT + 512 * VTS)       // 136192
#define SMEM_HALVES (OFF_PB + 8 * 16 * PS) // 145408
#define SMEM_BYTES  (SMEM_HALVES * 2)      // 290816 (< 320KB WGP LDS)

// Load a 16-half WMMA operand: halves 0-7 at p, halves 8-15 at p+16.
// Caller bakes in the lane-half (ks*8) offset per the CDNA5 A/B VGPR layout.
__device__ __forceinline__ h16 ld_op16(const _Float16* p) {
  union { h16 v; h8 h[2]; } u;
  u.h[0] = *(const h8*)(p);
  u.h[1] = *(const h8*)(p + 16);
  return u.v;
}

__device__ __forceinline__ f8 wmma_f16(h16 a, h16 b, f8 c) {
  // D = A(16x32) * B(32x16) + C, f32 accumulate
  return __builtin_amdgcn_wmma_f32_16x16x32_f16(
      /*neg_a=*/false, a, /*neg_b=*/false, b,
      /*c_mod=*/(short)0, c, /*reuse_a=*/false, /*reuse_b=*/false);
}

// ---------------- weight prep: f32 [K][N] -> f16 [N][K] ----------------
__global__ void prep_weights(const float* __restrict__ qkv_w,
                             const float* __restrict__ proj_w,
                             _Float16* __restrict__ qkvT,
                             _Float16* __restrict__ projT) {
  int i = blockIdx.x * blockDim.x + threadIdx.x;
  const int n1 = CDIM * C3;
  if (i < n1) {
    int k = i / C3, n = i % C3;
    qkvT[(size_t)n * CDIM + k] = (_Float16)qkv_w[i];
  } else if (i < n1 + CDIM * CDIM) {
    int j = i - n1;
    int k = j / CDIM, n = j % CDIM;
    projT[(size_t)n * CDIM + k] = (_Float16)proj_w[j];
  }
}

// ---------------- fused window attention: one block per window ----------------
__global__ void __launch_bounds__(256)
fused_window_attn(const float* __restrict__ x,
                  const float* __restrict__ qkv_b,
                  const float* __restrict__ proj_b,
                  const _Float16* __restrict__ qkvT,
                  const _Float16* __restrict__ projT,
                  float* __restrict__ out) {
  extern __shared__ _Float16 sm[];
  _Float16* xh = sm + OFF_XH;   // x in f16; later reused as attention output
  _Float16* qk = sm + OFF_QK;   // cols 0..511 = scaled Q, 512..1023 = K
  _Float16* vT = sm + OFF_VT;   // vT[col (h*32+d)][token]
  _Float16* pb = sm + OFF_PB;   // per-wave softmax-P staging

  const int b    = blockIdx.x;
  const int t    = threadIdx.x;
  const int wave = t >> 5;
  const int lane = t & 31;
  const int lh   = lane & 15;   // operand column/row within 16
  const int ks   = lane >> 4;   // lane-half selects K sub-block
  const float scale = 0.17677669529663687f; // 32^-0.5

  // ---- Phase 0: load x[b] (60x512 f32) -> f16 LDS, zero padded rows ----
  {
    const float4* xb = (const float4*)(x + (size_t)b * NA * CDIM);
    for (int i = t; i < NA * (CDIM / 4); i += 256) {
      float4 v = xb[i];
      int r = i >> 7, c = (i & 127) << 2;
      h4 o = {(_Float16)v.x, (_Float16)v.y, (_Float16)v.z, (_Float16)v.w};
      *(h4*)(xh + r * XS + c) = o;
    }
    for (int i = t; i < (NW - NA) * XS; i += 256)
      xh[NA * XS + i] = (_Float16)0.f;
  }
  __syncthreads();

  // ---- Phase 1: QKV GEMM (64x512 @ 512x1536) ----
  // 12 N-tiles per wave, register-blocked in 3 groups of 4: acc[m][n],
  // so each LDS A tile feeds 4 WMMAs and each B tile feeds 4 WMMAs.
  for (int g = 0; g < 3; ++g) {
    const int nt0 = wave * 12 + g * 4;
    f8 acc[4][4] = {};
    const _Float16* bbase = qkvT + (size_t)(nt0 * 16 + lh) * CDIM + ks * 8;
    for (int k = 0; k < 16; ++k) {
      h16 bm[4];
#pragma unroll
      for (int n = 0; n < 4; ++n) {
        const _Float16* bp = bbase + (size_t)n * 16 * CDIM + k * 32;
        __builtin_prefetch(bp + 32, 0, 3);      // next k-step, near cache
        bm[n] = ld_op16(bp);
      }
      const _Float16* abase = xh + lh * XS + k * 32 + ks * 8;
#pragma unroll
      for (int m = 0; m < 4; ++m) {
        h16 am = ld_op16(abase + m * 16 * XS);
#pragma unroll
        for (int n = 0; n < 4; ++n)
          acc[m][n] = wmma_f16(am, bm[n], acc[m][n]);
      }
    }
#pragma unroll
    for (int n = 0; n < 4; ++n) {
      const int ncol = (nt0 + n) * 16;
      const float bias = qkv_b[ncol + lh];
      if (ncol < CDIM) {                 // Q: (acc+b)*scale, row-major
#pragma unroll
        for (int m = 0; m < 4; ++m)
#pragma unroll
          for (int i = 0; i < 8; ++i)
            qk[(m * 16 + i + ks * 8) * QKS + ncol + lh] =
                (_Float16)((acc[m][n][i] + bias) * scale);
      } else if (ncol < 2 * CDIM) {      // K: acc+b, row-major
#pragma unroll
        for (int m = 0; m < 4; ++m)
#pragma unroll
          for (int i = 0; i < 8; ++i)
            qk[(m * 16 + i + ks * 8) * QKS + ncol + lh] =
                (_Float16)(acc[m][n][i] + bias);
      } else {                           // V: acc+b, TRANSPOSED store
        const int vrow = ncol - 2 * CDIM + lh;
#pragma unroll
        for (int m = 0; m < 4; ++m)
#pragma unroll
          for (int i = 0; i < 8; ++i)
            vT[vrow * VTS + (m * 16 + i + ks * 8)] =
                (_Float16)(acc[m][n][i] + bias);
      }
    }
  }
  __syncthreads();

  // ---- Phase 2: per-head attention (2 heads per wave) ----
  _Float16* pw = pb + wave * 16 * PS;
  for (int hi = 0; hi < 2; ++hi) {
    const int h = wave + hi * 8;
    for (int mt = 0; mt < 4; ++mt) {
      // scores S[16x64] = Qtile(16x32) @ K^T : one K=32 WMMA per 16x16 tile
      f8 s[4] = {};
      h16 qa = ld_op16(qk + (mt * 16 + lh) * QKS + h * HD + ks * 8);
#pragma unroll
      for (int nt = 0; nt < 4; ++nt) {
        h16 kb = ld_op16(qk + (nt * 16 + lh) * QKS + CDIM + h * HD + ks * 8);
        s[nt] = wmma_f16(qa, kb, s[nt]);
      }
      // mask cols >= 60, row softmax across 4 regs x 16 lanes (half-wave)
#pragma unroll
      for (int i = 0; i < 8; ++i) {
        float v0 = s[0][i], v1 = s[1][i], v2 = s[2][i], v3 = s[3][i];
        if (lh >= 12) v3 = -1e30f;     // key tokens 60..63 are padding
        float mx = fmaxf(fmaxf(v0, v1), fmaxf(v2, v3));
#pragma unroll
        for (int o = 1; o < 16; o <<= 1) mx = fmaxf(mx, __shfl_xor(mx, o, 32));
        v0 = __expf(v0 - mx); v1 = __expf(v1 - mx);
        v2 = __expf(v2 - mx); v3 = __expf(v3 - mx);
        float sum = v0 + v1 + v2 + v3;
#pragma unroll
        for (int o = 1; o < 16; o <<= 1) sum += __shfl_xor(sum, o, 32);
        const float inv = __builtin_amdgcn_rcpf(sum);
        const int r = i + ks * 8;      // local row in this 16-row block
        pw[r * PS +  0 + lh] = (_Float16)(v0 * inv);
        pw[r * PS + 16 + lh] = (_Float16)(v1 * inv);
        pw[r * PS + 32 + lh] = (_Float16)(v2 * inv);
        pw[r * PS + 48 + lh] = (_Float16)(v3 * inv);
      }
      // out[16x32] = P(16x64) @ V(64x32): 2 d-tiles x 2 K-steps
#pragma unroll
      for (int dt = 0; dt < 2; ++dt) {
        f8 o = {};
#pragma unroll
        for (int kk = 0; kk < 2; ++kk) {
          h16 pa = ld_op16(pw + lh * PS + kk * 32 + ks * 8);
          h16 vb = ld_op16(vT + (h * HD + dt * 16 + lh) * VTS + kk * 32 + ks * 8);
          o = wmma_f16(pa, vb, o);
        }
#pragma unroll
        for (int i = 0; i < 8; ++i)
          xh[(mt * 16 + i + ks * 8) * XS + h * HD + dt * 16 + lh] = (_Float16)o[i];
      }
    }
  }
  __syncthreads();

  // ---- Phase 3: output projection (64x512 @ 512x512) ----
  // 4 N-tiles per wave = one register-blocked group.
  float* yb = out + (size_t)b * NA * CDIM;
  {
    const int nt0 = wave * 4;
    f8 acc[4][4] = {};
    const _Float16* bbase = projT + (size_t)(nt0 * 16 + lh) * CDIM + ks * 8;
    for (int k = 0; k < 16; ++k) {
      h16 bm[4];
#pragma unroll
      for (int n = 0; n < 4; ++n) {
        const _Float16* bp = bbase + (size_t)n * 16 * CDIM + k * 32;
        __builtin_prefetch(bp + 32, 0, 3);
        bm[n] = ld_op16(bp);
      }
      const _Float16* abase = xh + lh * XS + k * 32 + ks * 8;
#pragma unroll
      for (int m = 0; m < 4; ++m) {
        h16 am = ld_op16(abase + m * 16 * XS);
#pragma unroll
        for (int n = 0; n < 4; ++n)
          acc[m][n] = wmma_f16(am, bm[n], acc[m][n]);
      }
    }
#pragma unroll
    for (int n = 0; n < 4; ++n) {
      const int ncol = (nt0 + n) * 16;
      const float bias = proj_b[ncol + lh];
#pragma unroll
      for (int m = 0; m < 4; ++m)
#pragma unroll
        for (int i = 0; i < 8; ++i) {
          const int r = m * 16 + i + ks * 8;
          if (r < NA) yb[(size_t)r * CDIM + ncol + lh] = acc[m][n][i] + bias;
        }
    }
  }
}

// ---------------- launch ----------------
extern "C" void kernel_launch(void* const* d_in, const int* in_sizes, int n_in,
                              void* d_out, int out_size, void* d_ws, size_t ws_size,
                              hipStream_t stream) {
  (void)in_sizes; (void)n_in; (void)out_size; (void)ws_size;
  const float* x      = (const float*)d_in[0];
  const float* qkv_w  = (const float*)d_in[1];
  const float* qkv_b  = (const float*)d_in[2];
  const float* proj_w = (const float*)d_in[3];
  const float* proj_b = (const float*)d_in[4];
  float* y = (float*)d_out;

  _Float16* qkvT = (_Float16*)d_ws;                    // [1536][512] f16
  _Float16* projT = qkvT + (size_t)C3 * CDIM;          // [512][512]  f16

  {
    const int total = CDIM * C3 + CDIM * CDIM;         // 1,048,576
    prep_weights<<<(total + 255) / 256, 256, 0, stream>>>(qkv_w, proj_w, qkvT, projT);
  }

  hipFuncSetAttribute((const void*)fused_window_attn,
                      hipFuncAttributeMaxDynamicSharedMemorySize, SMEM_BYTES);
  fused_window_attn<<<2048, 256, SMEM_BYTES, stream>>>(x, qkv_b, proj_b,
                                                       qkvT, projT, y);
}
